// EGNNLayer_11742440587289
// MI455X (gfx1250) — compile-verified, matrix operand order checked by reading, not run on hardware
//
#include <hip/hip_runtime.h>
#include <hip/hip_bf16.h>
#include <math.h>

#define NN   50000
#define EE   800000
#define FIN  128
#define FH   128
#define EDGE_TILES32 (EE/32)   // 25000 tiles of 32 edges
#define NODE_TILES   (NN/16)   // 3125 tiles of 16 nodes

typedef __attribute__((ext_vector_type(16))) __bf16 v16bf;
typedef __attribute__((ext_vector_type(8)))  __bf16 v8bf;
typedef __attribute__((ext_vector_type(8)))  float  v8f;

__device__ __forceinline__ float silu_f(float v) {
    return v / (1.0f + __expf(-v));
}

// Guaranteed single-instruction f32 atomic add (no-return, device scope).
__device__ __forceinline__ void atomAddG(float* p, float v) {
    asm volatile("global_atomic_add_f32 %0, %1, off scope:SCOPE_DEV"
                 :: "v"(p), "v"(v) : "memory");
}

__device__ __forceinline__ v16bf make_a(float4 l0, float4 l1, float4 h0, float4 h1) {
    v16bf a;
    a[0]=(__bf16)l0.x;  a[1]=(__bf16)l0.y;  a[2]=(__bf16)l0.z;  a[3]=(__bf16)l0.w;
    a[4]=(__bf16)l1.x;  a[5]=(__bf16)l1.y;  a[6]=(__bf16)l1.z;  a[7]=(__bf16)l1.w;
    a[8]=(__bf16)h0.x;  a[9]=(__bf16)h0.y;  a[10]=(__bf16)h0.z; a[11]=(__bf16)h0.w;
    a[12]=(__bf16)h1.x; a[13]=(__bf16)h1.y; a[14]=(__bf16)h1.z; a[15]=(__bf16)h1.w;
    return a;
}

__device__ __forceinline__ v16bf combine8(v8bf lo, v8bf hi) {
    v16bf a;
#pragma unroll
    for (int i = 0; i < 8; ++i) { a[i] = lo[i]; a[8+i] = hi[i]; }
    return a;
}

__device__ __forceinline__ v16bf gather_a(const float* px) {
    float4 l0 = *(const float4*)(px);
    float4 l1 = *(const float4*)(px + 4);
    float4 h0 = *(const float4*)(px + 16);
    float4 h1 = *(const float4*)(px + 20);
    return make_a(l0, l1, h0, h1);
}

// ---------------------------------------------------------------- zero ws
__global__ void zero_ws(float* p, int n) {
    int i = blockIdx.x * blockDim.x + threadIdx.x;
    int stride = gridDim.x * blockDim.x;
    for (; i < n; i += stride) p[i] = 0.f;
}

// -------------------------------------------------- weight fragment prep
// B-fragment layout for v_wmma_f32_16x16x32_bf16 (per 32x16 K×N tile):
// lane L holds column n = nt*16 + (L&15); its 16 bf16 elements e map to
// k = kt*32 + (e/8)*16 + (L/16)*8 + (e%8). Stored per-lane contiguous
// (32 bytes) so a wave loads one fragment with a single 32B vector load.
__global__ void prep_frags(const float* eW1, const float* eW2,
                           const float* nW1, const float* nW2,
                           __bf16* fE1, __bf16* fE2, __bf16* fN1, __bf16* fN2) {
    int t = blockIdx.x * blockDim.x + threadIdx.x;   // 0 .. 98303
    int fi   = t >> 9;       // global fragment id (192 total)
    int r    = t & 511;
    int lane = r >> 4;
    int e    = r & 15;
    const float* W; __bf16* dst; int lf;
    if      (fi <  64) { W = eW1; dst = fE1; lf = fi;       }
    else if (fi <  96) { W = eW2; dst = fE2; lf = fi - 64;  }
    else if (fi < 160) { W = nW1; dst = fN1; lf = fi - 96;  }
    else               { W = nW2; dst = fN2; lf = fi - 160; }
    int kt = lf >> 3, nt = lf & 7;
    int k = kt*32 + (e >> 3)*16 + (lane >> 4)*8 + (e & 7);
    int n = nt*16 + (lane & 15);
    dst[lf*512 + lane*16 + e] = (__bf16)W[k*FH + n];
}

// ---------------------------------------------------------- edge kernel
// 1 wave = 32 edges (2 M-tiles): each B fragment loaded once feeds 2 WMMAs.
// Fused edge MLP (bf16 WMMA) -> silu -> atomic segment-sum into agg,
// plus w_ij = silu(m@cW+cb) -> delta atomics. 4 waves per block.
__launch_bounds__(128)
__global__ void edge_kernel(const float* __restrict__ x,
                            const float* __restrict__ pos,
                            const long long* __restrict__ ei,
                            const float* __restrict__ eW1,
                            const float* __restrict__ eb1,
                            const float* __restrict__ eb2,
                            const float* __restrict__ cW,
                            const float* __restrict__ cb,
                            const __bf16* __restrict__ fE1,
                            const __bf16* __restrict__ fE2,
                            float* __restrict__ agg,
                            float* __restrict__ delta) {
    __shared__ __bf16 sm1[4][32*FH];      // per-wave bf16 m1 tile (8 KB each)
    __shared__ float  sdij[4][32], srnx[4][32], srny[4][32], srnz[4][32];
    __shared__ int    srow[4][32], scol[4][32];
    __shared__ float  swac[4][32];

    const int w    = threadIdx.x >> 5;
    const int lane = threadIdx.x & 31;
    const int lm   = lane & 15;
    const int h    = lane >> 4;
    const int tile = blockIdx.x * 4 + w;          // 32-edge tile

    // ---- per-edge metadata: all 32 lanes, one edge each
    {
        long long e = (long long)tile * 32 + lane;
        int r = (int)ei[e];
        int c = (int)ei[(long long)EE + e];
        float dx = pos[r*3+0] - pos[c*3+0];
        float dy = pos[r*3+1] - pos[c*3+1];
        float dz = pos[r*3+2] - pos[c*3+2];
        float d2 = dx*dx + dy*dy + dz*dz;
        float inv = 1.0f / (sqrtf(d2) + 1e-8f);
        sdij[w][lane] = d2;
        srnx[w][lane] = dx*inv; srny[w][lane] = dy*inv; srnz[w][lane] = dz*inv;
        srow[w][lane] = r; scol[w][lane] = c;
        swac[w][lane] = 0.f;
    }
    __builtin_amdgcn_wave_barrier();   // LDS is in-order per wave

    float d8[2][8];
#pragma unroll
    for (int mt = 0; mt < 2; ++mt)
#pragma unroll
        for (int v = 0; v < 8; ++v) d8[mt][v] = sdij[w][mt*16 + v + 8*h];
    const int rA[2] = { srow[w][lm], srow[w][16 + lm] };
    const int cA[2] = { scol[w][lm], scol[w][16 + lm] };

    // ---- layer 1 accumulators: bias + rank-1 distance term
    v8f acc[2][8];
#pragma unroll
    for (int nt = 0; nt < 8; ++nt) {
        int n = nt*16 + lm;
        float bias  = eb1[n];
        float wlast = eW1[256*FH + n];   // eW1 row for dij feature
#pragma unroll
        for (int mt = 0; mt < 2; ++mt)
#pragma unroll
            for (int v = 0; v < 8; ++v) acc[mt][nt][v] = bias + d8[mt][v]*wlast;
    }

    // ---- layer 1 GEMM: [32,256]@[256,128], K-tiles 0..3 = x[row], 4..7 = x[col]
#pragma unroll
    for (int kt = 0; kt < 8; ++kt) {
        v16bf a[2];
#pragma unroll
        for (int mt = 0; mt < 2; ++mt) {
            int r16 = (kt < 4) ? rA[mt] : cA[mt];
            a[mt] = gather_a(x + (long long)r16*FIN + (kt & 3)*32 + h*8);
        }
#pragma unroll
        for (int nt = 0; nt < 8; ++nt) {
            v16bf b = *(const v16bf*)(fE1 + (kt*8 + nt)*512 + lane*16);
            acc[0][nt] = __builtin_amdgcn_wmma_f32_16x16x32_bf16(
                false, a[0], false, b, (short)0, acc[0][nt], false, false);
            acc[1][nt] = __builtin_amdgcn_wmma_f32_16x16x32_bf16(
                false, a[1], false, b, (short)0, acc[1][nt], false, false);
        }
    }

    // ---- silu -> bf16 m1 tile in LDS (row-major [edge][feat])
#pragma unroll
    for (int mt = 0; mt < 2; ++mt)
#pragma unroll
        for (int nt = 0; nt < 8; ++nt)
#pragma unroll
            for (int v = 0; v < 8; ++v)
                sm1[w][(mt*16 + v + 8*h)*FH + nt*16 + lm] =
                    (__bf16)silu_f(acc[mt][nt][v]);
    __builtin_amdgcn_wave_barrier();

    // ---- layer 2 GEMM: [32,128]@[128,128]
    v8f acc2[2][8];
#pragma unroll
    for (int nt = 0; nt < 8; ++nt) {
        float bias = eb2[nt*16 + lm];
#pragma unroll
        for (int mt = 0; mt < 2; ++mt)
#pragma unroll
            for (int v = 0; v < 8; ++v) acc2[mt][nt][v] = bias;
    }
#pragma unroll
    for (int kt = 0; kt < 4; ++kt) {
        v16bf a[2];
#pragma unroll
        for (int mt = 0; mt < 2; ++mt) {
            const __bf16* mrow = &sm1[w][(mt*16 + lm)*FH];
            v8bf lo = *(const v8bf*)(mrow + kt*32 + h*8);
            v8bf hi = *(const v8bf*)(mrow + kt*32 + 16 + h*8);
            a[mt] = combine8(lo, hi);
        }
#pragma unroll
        for (int nt = 0; nt < 8; ++nt) {
            v16bf b = *(const v16bf*)(fE2 + (kt*8 + nt)*512 + lane*16);
            acc2[0][nt] = __builtin_amdgcn_wmma_f32_16x16x32_bf16(
                false, a[0], false, b, (short)0, acc2[0][nt], false, false);
            acc2[1][nt] = __builtin_amdgcn_wmma_f32_16x16x32_bf16(
                false, a[1], false, b, (short)0, acc2[1][nt], false, false);
        }
    }

    // ---- silu -> m2; scatter into agg; accumulate cW dot
    int rIdx[2][8];
#pragma unroll
    for (int mt = 0; mt < 2; ++mt)
#pragma unroll
        for (int v = 0; v < 8; ++v) rIdx[mt][v] = srow[w][mt*16 + v + 8*h];
    float p[2][8];
#pragma unroll
    for (int mt = 0; mt < 2; ++mt)
#pragma unroll
        for (int v = 0; v < 8; ++v) p[mt][v] = 0.f;

#pragma unroll
    for (int nt = 0; nt < 8; ++nt) {
        int n = nt*16 + lm;
        float cw = cW[n];
#pragma unroll
        for (int mt = 0; mt < 2; ++mt)
#pragma unroll
            for (int v = 0; v < 8; ++v) {
                float m2 = silu_f(acc2[mt][nt][v]);
                atomAddG(&agg[(long long)rIdx[mt][v]*FH + n], m2);
                p[mt][v] += m2 * cw;
            }
    }
#pragma unroll
    for (int mt = 0; mt < 2; ++mt)
#pragma unroll
        for (int v = 0; v < 8; ++v)
            __hip_atomic_fetch_add(&swac[w][mt*16 + v + 8*h], p[mt][v],
                                   __ATOMIC_RELAXED, __HIP_MEMORY_SCOPE_WORKGROUP);
    __builtin_amdgcn_wave_barrier();

    // all 32 lanes: one edge each
    {
        float wv = silu_f(swac[w][lane] + cb[0]);
        int r = srow[w][lane];
        atomAddG(&delta[r*3+0], srnx[w][lane]*wv);
        atomAddG(&delta[r*3+1], srny[w][lane]*wv);
        atomAddG(&delta[r*3+2], srnz[w][lane]*wv);
    }
}

// ---------------------------------------------------------- node kernel
// 1 wave = 16 nodes: h = silu([x|agg]@nW1 + nb1); x_out = x + h@nW2 + nb2
__launch_bounds__(256)
__global__ void node_kernel(const float* __restrict__ x,
                            const float* __restrict__ agg,
                            const float* __restrict__ nb1,
                            const float* __restrict__ nb2,
                            const __bf16* __restrict__ fN1,
                            const __bf16* __restrict__ fN2,
                            float* __restrict__ xout) {
    __shared__ __bf16 sh1[8][16*FH];
    const int w    = threadIdx.x >> 5;
    const int lane = threadIdx.x & 31;
    const int lm   = lane & 15;
    const int h    = lane >> 4;
    const int tile = blockIdx.x * 8 + w;
    if (tile >= NODE_TILES) return;

    const long long nodeRow = (long long)tile*16 + lm;

    v8f acc[8];
#pragma unroll
    for (int nt = 0; nt < 8; ++nt) {
        float b = nb1[nt*16 + lm];
#pragma unroll
        for (int v = 0; v < 8; ++v) acc[nt][v] = b;
    }
#pragma unroll
    for (int kt = 0; kt < 8; ++kt) {
        const float* src = (kt < 4)
            ? (x   + nodeRow*FIN + (kt & 3)*32 + h*8)
            : (agg + nodeRow*FH  + (kt & 3)*32 + h*8);
        v16bf a = gather_a(src);
#pragma unroll
        for (int nt = 0; nt < 8; ++nt) {
            v16bf b = *(const v16bf*)(fN1 + (kt*8 + nt)*512 + lane*16);
            acc[nt] = __builtin_amdgcn_wmma_f32_16x16x32_bf16(
                false, a, false, b, (short)0, acc[nt], false, false);
        }
    }
#pragma unroll
    for (int nt = 0; nt < 8; ++nt)
#pragma unroll
        for (int v = 0; v < 8; ++v)
            sh1[w][(v + 8*h)*FH + nt*16 + lm] = (__bf16)silu_f(acc[nt][v]);
    __builtin_amdgcn_wave_barrier();

    v8f acc2[8];
#pragma unroll
    for (int nt = 0; nt < 8; ++nt) {
        float b = nb2[nt*16 + lm];
#pragma unroll
        for (int v = 0; v < 8; ++v) acc2[nt][v] = b;
    }
    const __bf16* mrow = &sh1[w][lm*FH];
#pragma unroll
    for (int kt = 0; kt < 4; ++kt) {
        v8bf lo = *(const v8bf*)(mrow + kt*32 + h*8);
        v8bf hi = *(const v8bf*)(mrow + kt*32 + 16 + h*8);
        v16bf a = combine8(lo, hi);
#pragma unroll
        for (int nt = 0; nt < 8; ++nt) {
            v16bf b = *(const v16bf*)(fN2 + (kt*8 + nt)*512 + lane*16);
            acc2[nt] = __builtin_amdgcn_wmma_f32_16x16x32_bf16(
                false, a, false, b, (short)0, acc2[nt], false, false);
        }
    }
    // residual write-out
#pragma unroll
    for (int nt = 0; nt < 8; ++nt) {
        int n = nt*16 + lm;
#pragma unroll
        for (int v = 0; v < 8; ++v) {
            long long node = (long long)tile*16 + v + 8*h;
            xout[node*FIN + n] = x[node*FIN + n] + acc2[nt][v];
        }
    }
}

// ------------------------------------------------------------ pos kernel
__global__ void pos_kernel(const float* pos, const float* delta, float* posout) {
    int i = blockIdx.x * blockDim.x + threadIdx.x;
    if (i < NN*3) posout[i] = pos[i] + delta[i];
}

// --------------------------------------------------------------- launch
extern "C" void kernel_launch(void* const* d_in, const int* in_sizes, int n_in,
                              void* d_out, int out_size, void* d_ws, size_t ws_size,
                              hipStream_t stream) {
    const float*     x   = (const float*)d_in[0];
    const float*     pos = (const float*)d_in[1];
    const long long* ei  = (const long long*)d_in[2];
    const float* eW1 = (const float*)d_in[3];
    const float* eb1 = (const float*)d_in[4];
    const float* eW2 = (const float*)d_in[5];
    const float* eb2 = (const float*)d_in[6];
    const float* nW1 = (const float*)d_in[7];
    const float* nb1 = (const float*)d_in[8];
    const float* nW2 = (const float*)d_in[9];
    const float* nb2 = (const float*)d_in[10];
    const float* cW  = (const float*)d_in[11];
    const float* cb  = (const float*)d_in[12];

    float* xout   = (float*)d_out;
    float* posout = xout + (size_t)NN*FIN;

    float* agg   = (float*)d_ws;                 // [N,128]
    float* delta = agg + (size_t)NN*FH;          // [N,3]
    __bf16* fE1 = (__bf16*)(delta + (size_t)NN*3);
    __bf16* fE2 = fE1 + 64*512;
    __bf16* fN1 = fE2 + 32*512;
    __bf16* fN2 = fN1 + 64*512;

    zero_ws<<<4096, 256, 0, stream>>>(agg, NN*FH + NN*3);
    prep_frags<<<384, 256, 0, stream>>>(eW1, eW2, nW1, nW2, fE1, fE2, fN1, fN2);
    edge_kernel<<<EDGE_TILES32/4, 128, 0, stream>>>(x, pos, ei, eW1, eb1, eb2,
                                                    cW, cb, fE1, fE2, agg, delta);
    node_kernel<<<(NODE_TILES + 7)/8, 256, 0, stream>>>(x, agg, nb1, nb2,
                                                        fN1, fN2, xout);
    pos_kernel<<<(NN*3 + 255)/256, 256, 0, stream>>>(pos, delta, posout);
}
